// MultiHeadAttention_13280038879884
// MI455X (gfx1250) — matrix operand order repaired and checked by reading, not compiled, not run on hardware
//
#include <hip/hip_runtime.h>
#include <hip/hip_bf16.h>

// ---- problem constants (match reference) ----
#define BB   2
#define SS   2048
#define DD   1024
#define HH   16
#define HDD  64
// ln(100000)
#define LN_ROPE_BASE 11.512925464970229f

typedef __attribute__((ext_vector_type(16))) __bf16 v16bf;
typedef __attribute__((ext_vector_type(8)))  float  v8f;

// ---------------------------------------------------------------------------
// gfx1250 async global->LDS copy helpers (ASYNCcnt-tracked, bypass VGPRs)
// ---------------------------------------------------------------------------
__device__ __forceinline__ unsigned ldsOffset(const void* p) {
  // addrspace(3) pointers are LDS byte offsets on amdgcn
  return (unsigned)(unsigned long long)(const __attribute__((address_space(3))) void*)p;
}

__device__ __forceinline__ void async_b128(unsigned ldsoff, const __bf16* g) {
  asm volatile("global_load_async_to_lds_b128 %0, %1, off"
               :: "v"(ldsoff), "v"(g) : "memory");
}

__device__ __forceinline__ void wait_async0() {
  asm volatile("s_wait_asynccnt 0x0" ::: "memory");
}

// ---------------------------------------------------------------------------
// Kernel 0a: f32 -> bf16 elementwise conversion
// ---------------------------------------------------------------------------
__global__ __launch_bounds__(256) void cvt_f32_bf16(const float* __restrict__ in,
                                                    __bf16* __restrict__ out,
                                                    size_t n) {
  size_t i = (size_t)blockIdx.x * blockDim.x + threadIdx.x;
  if (i < n) out[i] = (__bf16)in[i];
}

// ---------------------------------------------------------------------------
// Kernel 0b: f32 [R][C] -> bf16 transposed [C][R] (weights, done once)
// ---------------------------------------------------------------------------
__global__ __launch_bounds__(256) void transpose_cvt_f32_bf16(const float* __restrict__ in,
                                                              __bf16* __restrict__ out,
                                                              int R, int C) {
  int i = blockIdx.x * 256 + threadIdx.x;
  int r = i / C, c = i % C;
  out[(size_t)c * R + r] = (__bf16)in[i];
}

// ---------------------------------------------------------------------------
// Kernel 1: bf16 WMMA GEMM  C[MxN] = A[MxK] * Bt[NxK]^T  (f32 out)
// Block tile 128x128, BK=32, 8 waves (4xM, 2xN), 32x64 per wave.
// Double-buffered async global->LDS staging overlapped with WMMA.
// ---------------------------------------------------------------------------
#define BM 128
#define BN 128
#define BK 32

__global__ __launch_bounds__(256) void gemm_bf16(const __bf16* __restrict__ A,
                                                 const __bf16* __restrict__ Bt,
                                                 float* __restrict__ C,
                                                 int M, int N, int K) {
  __shared__ __bf16 ldsA[2][BM * BK];   // [row][k]
  __shared__ __bf16 ldsB[2][BN * BK];   // [col][k]

  const int tid  = threadIdx.x;
  const int wave = tid >> 5;
  const int lane = tid & 31;
  const int half = lane >> 4;
  const int l16  = lane & 15;

  const int tileM0 = blockIdx.y * BM;
  const int tileN0 = blockIdx.x * BN;
  const int waveRow = (wave & 3) * 32;
  const int waveCol = (wave >> 2) * 64;

  // per-thread staging coordinates: 16 contiguous bf16 of one row
  const int sr = tid >> 1;            // 0..127 (row of A / col-row of Bt)
  const int sc = (tid & 1) * 16;      // 0 or 16 within BK

  v8f acc[2][4];
#pragma unroll
  for (int i = 0; i < 2; i++)
#pragma unroll
    for (int j = 0; j < 4; j++)
#pragma unroll
      for (int r = 0; r < 8; r++) acc[i][j][r] = 0.0f;

  const int nk = K / BK;
  auto issue = [&](int kt, int buf) {
    const __bf16* asrc = A  + (size_t)(tileM0 + sr) * K + kt * BK + sc;
    const __bf16* bsrc = Bt + (size_t)(tileN0 + sr) * K + kt * BK + sc;
    unsigned ad = ldsOffset(&ldsA[buf][sr * BK + sc]);
    unsigned bd = ldsOffset(&ldsB[buf][sr * BK + sc]);
    async_b128(ad,      asrc);
    async_b128(ad + 16, asrc + 8);
    async_b128(bd,      bsrc);
    async_b128(bd + 16, bsrc + 8);
  };

  issue(0, 0);
  for (int kt = 0; kt < nk; kt++) {
    wait_async0();           // this wave's copies for tile kt landed
    __syncthreads();         // -> everyone's copies landed
    if (kt + 1 < nk) issue(kt + 1, (kt + 1) & 1);   // overlap next copies
    const __bf16* la = &ldsA[kt & 1][0];
    const __bf16* lb = &ldsB[kt & 1][0];

#pragma unroll
    for (int i = 0; i < 2; i++) {
      const int arow = waveRow + i * 16 + l16;
      // A fragment 16x32: elems 0..7 = K[half*8..], 8..15 = K[16+half*8..]
      v16bf af;
      *(uint4*)(&af)     = *(const uint4*)(la + arow * BK + half * 8);
      *((uint4*)(&af)+1) = *(const uint4*)(la + arow * BK + 16 + half * 8);
#pragma unroll
      for (int j = 0; j < 4; j++) {
        const int bcol = waveCol + j * 16 + l16;
        // B fragment 32x16: elems 0..7 = K[half*8..], 8..15 = K[16+half*8..]
        v16bf bf_;
        *(uint4*)(&bf_)     = *(const uint4*)(lb + bcol * BK + half * 8);
        *((uint4*)(&bf_)+1) = *(const uint4*)(lb + bcol * BK + 16 + half * 8);
        acc[i][j] = __builtin_amdgcn_wmma_f32_16x16x32_bf16(
            false, af, false, bf_, (short)0, acc[i][j], false, false);
      }
    }
  }

#pragma unroll
  for (int i = 0; i < 2; i++) {
#pragma unroll
    for (int j = 0; j < 4; j++) {
      const int col     = tileN0 + waveCol + j * 16 + l16;
      const int rowBase = tileM0 + waveRow + i * 16 + half * 8;
#pragma unroll
      for (int r = 0; r < 8; r++)
        C[(size_t)(rowBase + r) * N + col] = acc[i][j][r];
    }
  }
}

// ---------------------------------------------------------------------------
// Kernel 2: RoPE + RMS-norm + head split.
//   rope_en=1: [B,S,D] f32 -> [B,H,S,HD] bf16 (q,k)
//   rope_en=0: [B,S,D] f32 -> [B,H,HD,S] bf16 (v, head-major for attention)
// One wave per (b,s,h) row of 64: lane holds (x[lane], x[lane+32]).
// ---------------------------------------------------------------------------
__global__ __launch_bounds__(256) void rope_rms_split(const float* __restrict__ x,
                                                      __bf16* __restrict__ out,
                                                      int rope_en) {
  const int gwave = (int)((blockIdx.x * blockDim.x + threadIdx.x) >> 5);
  const int lane  = threadIdx.x & 31;
  const int h  = gwave % HH;
  const int bs = gwave / HH;            // b*S + s
  const int s  = bs % SS;
  const int b  = bs / SS;

  const float* row = x + (size_t)bs * DD + h * HDD;
  float x1 = row[lane];
  float x2 = row[lane + 32];

  if (rope_en) {
    // gamma = s * ROPE_BASE^(-lane/32); cos/sin rounded through bf16
    float g  = (float)s * __expf(-(float)lane * (LN_ROPE_BASE / 32.0f));
    float c  = (float)(__bf16)cosf(g);
    float sn = (float)(__bf16)sinf(g);
    float y1 =  x1 * c + x2 * sn;
    float y2 = -x1 * sn + x2 * c;
    float ss = y1 * y1 + y2 * y2;
#pragma unroll
    for (int o = 16; o >= 1; o >>= 1) ss += __shfl_xor(ss, o, 32);
    float inv = 1.0f / sqrtf(ss * (1.0f / (float)HDD) + 1e-9f);
    __bf16* orow = out + (((size_t)(b * HH + h) * SS + s) * HDD);
    orow[lane]      = (__bf16)(y1 * inv);
    orow[lane + 32] = (__bf16)(y2 * inv);
  } else {
    // v: head-major transpose [B,H,HD,S]
    __bf16* ob = out + (size_t)(b * HH + h) * HDD * SS;
    ob[(size_t)lane        * SS + s] = (__bf16)x1;
    ob[(size_t)(lane + 32) * SS + s] = (__bf16)x2;
  }
}

// ---------------------------------------------------------------------------
// Kernel 3: causal flash-attention. Grid (S/64, B*H), 128 threads (4 waves),
// each wave owns 16 query rows; 64-key K/V tiles double-buffered via async
// global->LDS copies overlapped with the WMMA pipeline.
// ---------------------------------------------------------------------------
__global__ __launch_bounds__(128) void flash_attn(const __bf16* __restrict__ qh,
                                                  const __bf16* __restrict__ kh,
                                                  const __bf16* __restrict__ vht,
                                                  float* __restrict__ vals) {
  const int bh = blockIdx.y;
  const int h  = bh % HH;
  const int b  = bh / HH;
  const int qt = blockIdx.x;                 // 64-query tile index

  const int tid  = threadIdx.x;
  const int wave = tid >> 5;
  const int lane = tid & 31;
  const int half = lane >> 4;
  const int l16  = lane & 15;

  const __bf16* Qp = qh  + ((size_t)bh * SS + (size_t)qt * 64) * HDD;
  const __bf16* Kp = kh  + (size_t)bh * SS * HDD;     // [key][hd]
  const __bf16* Vt = vht + (size_t)bh * HDD * SS;     // [hd][key]

  __shared__ __bf16 ldsK[2][64 * 72];        // [key][hd], padded
  __shared__ __bf16 ldsVt[2][64 * 72];       // [hd][key], padded
  __shared__ __bf16 ldsP[4][16 * 64];        // per-wave probs tile

  // staging coords: each thread moves half a 64-elem row (32 bf16 = 4 x b128)
  const int sr = tid >> 1;                   // 0..63 (key row of K / hd row of Vt)
  const int sc = (tid & 1) * 32;             // element offset within row

  auto issueKV = [&](int kb, int buf) {
    const __bf16* ks = Kp + ((size_t)kb * 64 + sr) * HDD + sc;
    const __bf16* vs = Vt + (size_t)sr * SS + kb * 64 + sc;
    unsigned kd = ldsOffset(&ldsK[buf][sr * 72 + sc]);
    unsigned vd = ldsOffset(&ldsVt[buf][sr * 72 + sc]);
#pragma unroll
    for (int c = 0; c < 4; c++) {
      async_b128(kd + c * 16, ks + c * 8);
      async_b128(vd + c * 16, vs + c * 8);
    }
  };

  // Q fragments (16 rows x 64 hd = 2 A-frags), loaded once from global
  v16bf qf[2];
  {
    const __bf16* qrow = Qp + (size_t)(wave * 16 + l16) * HDD;
#pragma unroll
    for (int j = 0; j < 2; j++) {
      *(uint4*)(&qf[j])     = *(const uint4*)(qrow + j * 32 + half * 8);
      *((uint4*)(&qf[j])+1) = *(const uint4*)(qrow + j * 32 + 16 + half * 8);
    }
  }

  v8f oacc[4];
#pragma unroll
  for (int j = 0; j < 4; j++)
#pragma unroll
    for (int r = 0; r < 8; r++) oacc[j][r] = 0.0f;

  float m[8], l[8];
#pragma unroll
  for (int r = 0; r < 8; r++) { m[r] = -3.0e38f; l[r] = 0.0f; }

  const int qrow0 = qt * 64 + wave * 16 + half * 8;
  const int nkb   = qt + 1;                  // causal: key blocks 0..qt
  const float scale = 0.125f;                // 1/sqrt(64)

  issueKV(0, 0);
  for (int kb = 0; kb < nkb; kb++) {
    wait_async0();
    __syncthreads();
    if (kb + 1 < nkb) issueKV(kb + 1, (kb + 1) & 1);
    const __bf16* lk = &ldsK[kb & 1][0];
    const __bf16* lv = &ldsVt[kb & 1][0];

    // ---- logits: 4 tiles of 16 keys, contraction over HD=64 (2 WMMA) ----
    v8f sacc[4];
#pragma unroll
    for (int t = 0; t < 4; t++) {
#pragma unroll
      for (int r = 0; r < 8; r++) sacc[t][r] = 0.0f;
#pragma unroll
      for (int j = 0; j < 2; j++) {
        v16bf bf_;
        const __bf16* kp = lk + (t * 16 + l16) * 72 + j * 32 + half * 8;
        *(uint4*)(&bf_)     = *(const uint4*)(kp);
        *((uint4*)(&bf_)+1) = *(const uint4*)(kp + 16);
        sacc[t] = __builtin_amdgcn_wmma_f32_16x16x32_bf16(
            false, qf[j], false, bf_, (short)0, sacc[t], false, false);
      }
    }

    // ---- online softmax update (rows spread as half*8+r, cols per lane) ----
    float corr[8];
#pragma unroll
    for (int r = 0; r < 8; r++) {
      float mx = -3.0e38f;
#pragma unroll
      for (int t = 0; t < 4; t++) {
        const int key = kb * 64 + t * 16 + l16;
        float v = sacc[t][r] * scale;
        v = (key <= qrow0 + r) ? v : -1.0e9f;   // causal mask (matches ref)
        sacc[t][r] = v;
        mx = fmaxf(mx, v);
      }
#pragma unroll
      for (int o = 1; o < 16; o <<= 1) mx = fmaxf(mx, __shfl_xor(mx, o, 32));
      const float mnew = fmaxf(m[r], mx);
      const float c    = __expf(m[r] - mnew);
      m[r] = mnew;
      float rs = 0.0f;
#pragma unroll
      for (int t = 0; t < 4; t++) {
        const float p = __expf(sacc[t][r] - mnew);
        sacc[t][r] = p;
        rs += p;
      }
#pragma unroll
      for (int o = 1; o < 16; o <<= 1) rs += __shfl_xor(rs, o, 32);
      l[r] = l[r] * c + rs;
      corr[r] = c;
    }
#pragma unroll
    for (int j = 0; j < 4; j++)
#pragma unroll
      for (int r = 0; r < 8; r++) oacc[j][r] *= corr[r];

    // ---- stage probs (bf16) to per-wave LDS, reload as A-fragments ----
    __bf16* pw = &ldsP[wave][0];
#pragma unroll
    for (int t = 0; t < 4; t++)
#pragma unroll
      for (int r = 0; r < 8; r++)
        pw[(half * 8 + r) * 64 + t * 16 + l16] = (__bf16)sacc[t][r];
    // same-wave LDS ops are in-order; explicit split-counter wait for safety
    asm volatile("s_wait_dscnt 0x0" ::: "memory");

    v16bf pf[2];
#pragma unroll
    for (int j = 0; j < 2; j++) {
      const __bf16* pr = pw + l16 * 64 + j * 32;
      *(uint4*)(&pf[j])     = *(const uint4*)(pr + half * 8);
      *((uint4*)(&pf[j])+1) = *(const uint4*)(pr + 16 + half * 8);
    }
    // ---- P(16x64) @ V(64x64): 4 hd tiles x 2 K-chunks ----
#pragma unroll
    for (int t = 0; t < 4; t++) {
#pragma unroll
      for (int j = 0; j < 2; j++) {
        v16bf bf_;
        const __bf16* vp = lv + (t * 16 + l16) * 72 + j * 32 + half * 8;
        *(uint4*)(&bf_)     = *(const uint4*)(vp);
        *((uint4*)(&bf_)+1) = *(const uint4*)(vp + 16);
        oacc[t] = __builtin_amdgcn_wmma_f32_16x16x32_bf16(
            false, pf[j], false, bf_, (short)0, oacc[t], false, false);
      }
    }
  }

  // ---- epilogue: normalize and scatter to vals[B,S,D] (d = h*64 + hd) ----
#pragma unroll
  for (int t = 0; t < 4; t++) {
    const int hd = t * 16 + l16;
#pragma unroll
    for (int r = 0; r < 8; r++) {
      const int s = qt * 64 + wave * 16 + half * 8 + r;
      vals[((size_t)b * SS + s) * DD + h * HDD + hd] = oacc[t][r] / l[r];
    }
  }
}

// ---------------------------------------------------------------------------
// Host-side orchestration
// ---------------------------------------------------------------------------
extern "C" void kernel_launch(void* const* d_in, const int* in_sizes, int n_in,
                              void* d_out, int out_size, void* d_ws, size_t ws_size,
                              hipStream_t stream) {
  (void)in_sizes; (void)n_in; (void)out_size; (void)ws_size;
  const float* Q  = (const float*)d_in[0];
  const float* K  = (const float*)d_in[1];
  const float* V  = (const float*)d_in[2];
  /* d_in[3] = mask: causality computed analytically */
  const float* Wq = (const float*)d_in[4];
  const float* Wk = (const float*)d_in[5];
  const float* Wv = (const float*)d_in[6];
  const float* Wo = (const float*)d_in[7];
  float* out = (float*)d_out;

  char* ws = (char*)d_ws;
  const size_t MiB = 1ull << 20;
  __bf16* Qbf = (__bf16*)(ws + 0 * MiB);     // 8 MiB each activation
  __bf16* Kbf = (__bf16*)(ws + 8 * MiB);
  __bf16* Vbf = (__bf16*)(ws + 16 * MiB);
  __bf16* Wqt = (__bf16*)(ws + 24 * MiB);    // 2 MiB each weight (transposed)
  __bf16* Wkt = (__bf16*)(ws + 26 * MiB);
  __bf16* Wvt = (__bf16*)(ws + 28 * MiB);
  __bf16* Wot = (__bf16*)(ws + 30 * MiB);
  float*  Qf  = (float*)(ws + 32 * MiB);     // 16 MiB each projection
  float*  Kf  = (float*)(ws + 48 * MiB);
  float*  Vf  = (float*)(ws + 64 * MiB);
  __bf16* qh  = (__bf16*)(ws + 80 * MiB);    // 8 MiB  [B,H,S,HD]
  __bf16* kh  = (__bf16*)(ws + 88 * MiB);    // 8 MiB  [B,H,S,HD]
  __bf16* vht = (__bf16*)(ws + 96 * MiB);    // 8 MiB  [B,H,HD,S]
  float*  vals  = Qf;                        // reuse (Qf dead after rope)
  __bf16* valsb = (__bf16*)Kf;               // reuse (Kf dead after rope)

  const size_t nX = (size_t)BB * SS * DD;    // 4M activation elems
  const size_t nW = (size_t)DD * DD;         // 1M weight elems
  const int M = BB * SS;                     // 4096

  // 0) convert activations to bf16; weights to transposed bf16
  cvt_f32_bf16<<<dim3((unsigned)(nX / 256)), 256, 0, stream>>>(Q, Qbf, nX);
  cvt_f32_bf16<<<dim3((unsigned)(nX / 256)), 256, 0, stream>>>(K, Kbf, nX);
  cvt_f32_bf16<<<dim3((unsigned)(nX / 256)), 256, 0, stream>>>(V, Vbf, nX);
  transpose_cvt_f32_bf16<<<dim3((unsigned)(nW / 256)), 256, 0, stream>>>(Wq, Wqt, DD, DD);
  transpose_cvt_f32_bf16<<<dim3((unsigned)(nW / 256)), 256, 0, stream>>>(Wk, Wkt, DD, DD);
  transpose_cvt_f32_bf16<<<dim3((unsigned)(nW / 256)), 256, 0, stream>>>(Wv, Wvt, DD, DD);
  transpose_cvt_f32_bf16<<<dim3((unsigned)(nW / 256)), 256, 0, stream>>>(Wo, Wot, DD, DD);

  // 1) projections (WMMA GEMMs, async-staged)
  dim3 gg(DD / BN, M / BM);
  gemm_bf16<<<gg, 256, 0, stream>>>(Qbf, Wqt, Qf, M, DD, DD);
  gemm_bf16<<<gg, 256, 0, stream>>>(Kbf, Wkt, Kf, M, DD, DD);
  gemm_bf16<<<gg, 256, 0, stream>>>(Vbf, Wvt, Vf, M, DD, DD);

  // 2) RoPE + RMS (q,k) / head-major reshape (v)
  const unsigned ropeBlocks = (unsigned)((size_t)BB * SS * HH * 32 / 256);
  rope_rms_split<<<ropeBlocks, 256, 0, stream>>>(Qf, qh, 1);
  rope_rms_split<<<ropeBlocks, 256, 0, stream>>>(Kf, kh, 1);
  rope_rms_split<<<ropeBlocks, 256, 0, stream>>>(Vf, vht, 0);

  // 3) causal flash attention (WMMA, async double-buffered K/V)
  flash_attn<<<dim3(SS / 64, BB * HH), 128, 0, stream>>>(qh, kh, vht, vals);

  // 4) output projection (WMMA GEMM) -> f32 result
  cvt_f32_bf16<<<dim3((unsigned)(nX / 256)), 256, 0, stream>>>(vals, valsb, nX);
  gemm_bf16<<<gg, 256, 0, stream>>>(valsb, Wot, out, M, DD, DD);
}
